// PartialFC_6786048328413
// MI455X (gfx1250) — matrix operand order repaired and checked by reading, not compiled, not run on hardware
//
#include <hip/hip_runtime.h>
#include <hip/hip_bf16.h>
#include <math.h>

typedef __attribute__((ext_vector_type(16))) _Float16 v16h;
typedef __attribute__((ext_vector_type(8)))  _Float16 v8h;
typedef __attribute__((ext_vector_type(2)))  _Float16 v2h;
typedef __attribute__((ext_vector_type(8)))  float    v8f;

#define B_SZ   256
#define D_SZ   512
#define C_SZ   250000
#define WAVES  8
#define NGROUPS 4
#define COLS_PER_WG (WAVES * 16 * NGROUPS)              // 512
#define NCG   ((C_SZ + COLS_PER_WG - 1) / COLS_PER_WG)  // 489
#define M_PER_WG 128
#define MTILES   (M_PER_WG / 16)                        // 8

__constant__ const float kScale  = 30.0f;
__constant__ const float kMargin = 0.2f;

// ---------------- kernel 1: normalize features, emit f16 A (row-major) -------
__global__ void k_norm_feat(const float* __restrict__ f,
                            _Float16* __restrict__ Af,
                            float* __restrict__ frnorm) {
    int row = blockIdx.x;            // 256 rows
    int t   = threadIdx.x;           // 256 threads, 2 elems each
    float2 v = *(const float2*)(f + (size_t)row * D_SZ + 2 * t);
    __shared__ float red[256];
    red[t] = v.x * v.x + v.y * v.y;
    __syncthreads();
    for (int s = 128; s > 0; s >>= 1) {
        if (t < s) red[t] += red[t + s];
        __syncthreads();
    }
    float rn = 1.0f / fmaxf(sqrtf(red[0]), 1e-12f);
    if (t == 0) frnorm[row] = rn;
    v2h h; h.x = (_Float16)(v.x * rn); h.y = (_Float16)(v.y * rn);
    *(v2h*)(Af + (size_t)row * D_SZ + 2 * t) = h;
}

// ---------------- kernel 2: exact f32 target cosine --------------------------
__global__ void k_target_cos(const float* __restrict__ f,
                             const int* __restrict__ tgt,
                             const float* __restrict__ w,
                             const float* __restrict__ frnorm,
                             float* __restrict__ cos_t) {
    int i = blockIdx.x;              // sample
    int t = threadIdx.x;             // 256 threads, 2 elems each
    int cls = tgt[i];
    const float* wr = w + (size_t)cls * D_SZ;
    float2 fv = *(const float2*)(f + (size_t)i * D_SZ + 2 * t);
    float2 wv = *(const float2*)(wr + 2 * t);
    __shared__ float rd[256], rs[256];
    rd[t] = fv.x * wv.x + fv.y * wv.y;
    rs[t] = wv.x * wv.x + wv.y * wv.y;
    __syncthreads();
    for (int s = 128; s > 0; s >>= 1) {
        if (t < s) { rd[t] += rd[t + s]; rs[t] += rs[t + s]; }
        __syncthreads();
    }
    if (t == 0)
        cos_t[i] = rd[0] * frnorm[i] / fmaxf(sqrtf(rs[0]), 1e-12f);
}

// ---------------- kernel 3: streaming WMMA GEMM + sum-exp --------------------
// Loop order: kc outer, m-tile inner.  One B fragment (8 VGPRs) per kc serves
// 8 independent accumulator chains (64 VGPRs) -> low register pressure, no
// serial WMMA dependence, weight still streamed exactly once from HBM.
__global__ __launch_bounds__(256, 2)
void k_main(const _Float16* __restrict__ Af,
            const float* __restrict__ w,
            float* __restrict__ gpart) {
    __shared__ _Float16 Alds[M_PER_WG * D_SZ];   // 128 KB f16 feature tile
    __shared__ float    wacc[WAVES][M_PER_WG];   // per-wave row partials

    const int mb   = blockIdx.x & 1;             // batch half
    const int cg   = blockIdx.x >> 1;            // class supertile
    const int t    = threadIdx.x;
    const int wave = t >> 5;
    const int lane = t & 31;
    const int col  = lane & 15;
    const int hi   = lane >> 4;                  // 0: lanes 0-15, 1: lanes 16-31

    // Stage A tile (rows mb*128 .. +128) into LDS: 8192 uint4
    {
        const uint4* src = (const uint4*)(Af + (size_t)mb * M_PER_WG * D_SZ);
        uint4* dst = (uint4*)Alds;
        for (int k = t; k < M_PER_WG * D_SZ / 8; k += 256) dst[k] = src[k];
        for (int k = t; k < WAVES * M_PER_WG; k += 256) ((float*)wacc)[k] = 0.0f;
    }
    __syncthreads();

    // per-lane LDS base for A fragments:
    // lanes 0-15 take K [0:16); lanes 16-31 take [16:32)
    const _Float16* aroot = &Alds[(size_t)col * D_SZ + hi * 16];

    #pragma unroll 1
    for (int grp = 0; grp < NGROUPS; ++grp) {
        const int  jbase = cg * COLS_PER_WG + grp * (WAVES * 16) + wave * 16;
        const int  j     = jbase + col;
        const bool valid = (j < C_SZ);
        const size_t jr  = (size_t)(valid ? j : 0) * D_SZ;
        // B layout: lanes 0-15 hold K 0..15 of each 32-chunk, lanes 16-31 hold K 16..31
        const float* wp = w + jr + hi * 16;

        // prefetch next group's weight rows (global_prefetch_b8)
        if (grp < NGROUPS - 1) {
            int jn = jbase + WAVES * 16 + col;
            const float* nwp = w + (size_t)((jn < C_SZ) ? jn : 0) * D_SZ;
            __builtin_prefetch((const void*)nwp, 0, 0);
            __builtin_prefetch((const void*)(nwp + 256), 0, 0);
        }

        float ssq = 0.0f;
        v8f c0 = {}, c1 = {}, c2 = {}, c3 = {};
        v8f c4 = {}, c5 = {}, c6 = {}, c7 = {};

        #pragma unroll 1
        for (int kc = 0; kc < 16; ++kc) {
            // one B fragment: this lane's 16 weight f32 -> 16 f16
            float4 a0 = *(const float4*)(wp + kc * 32 + 0);
            float4 a1 = *(const float4*)(wp + kc * 32 + 4);
            float4 a2 = *(const float4*)(wp + kc * 32 + 8);
            float4 a3 = *(const float4*)(wp + kc * 32 + 12);
            ssq += a0.x*a0.x + a0.y*a0.y + a0.z*a0.z + a0.w*a0.w
                 + a1.x*a1.x + a1.y*a1.y + a1.z*a1.z + a1.w*a1.w
                 + a2.x*a2.x + a2.y*a2.y + a2.z*a2.z + a2.w*a2.w
                 + a3.x*a3.x + a3.y*a3.y + a3.z*a3.z + a3.w*a3.w;
            v16h b;
            b[0]=(_Float16)a0.x;  b[1]=(_Float16)a0.y;  b[2]=(_Float16)a0.z;  b[3]=(_Float16)a0.w;
            b[4]=(_Float16)a1.x;  b[5]=(_Float16)a1.y;  b[6]=(_Float16)a1.z;  b[7]=(_Float16)a1.w;
            b[8]=(_Float16)a2.x;  b[9]=(_Float16)a2.y;  b[10]=(_Float16)a2.z; b[11]=(_Float16)a2.w;
            b[12]=(_Float16)a3.x; b[13]=(_Float16)a3.y; b[14]=(_Float16)a3.z; b[15]=(_Float16)a3.w;

            const _Float16* ak = aroot + kc * 32;
            #pragma unroll
            for (int mt = 0; mt < MTILES; ++mt) {
                const _Float16* ap = ak + (size_t)mt * 16 * D_SZ;
                v8h lo = *(const v8h*)ap;
                v8h hh = *(const v8h*)(ap + 8);
                v16h a = __builtin_shufflevector(lo, hh,
                        0,1,2,3,4,5,6,7,8,9,10,11,12,13,14,15);
                v8f& c = (mt==0)?c0:(mt==1)?c1:(mt==2)?c2:(mt==3)?c3:
                         (mt==4)?c4:(mt==5)?c5:(mt==6)?c6:c7;
                c = __builtin_amdgcn_wmma_f32_16x16x32_f16(
                        false, a, false, b, (short)0, c, false, false);
            }
        }

        // lanes l and l+16 cover complementary K halves of the same column
        ssq += __shfl_xor(ssq, 16);
        const float sc = kScale / fmaxf(sqrtf(ssq), 1e-12f);  // fold 1/||w|| into scale

        // c[r] on this lane = row (mt*16 + r + 8*hi), column j.  Sum exp over 16 cols.
        #pragma unroll
        for (int mt = 0; mt < MTILES; ++mt) {
            const v8f& c = (mt==0)?c0:(mt==1)?c1:(mt==2)?c2:(mt==3)?c3:
                           (mt==4)?c4:(mt==5)?c5:(mt==6)?c6:c7;
            #pragma unroll
            for (int r = 0; r < 8; ++r) {
                float v = valid ? __expf(sc * c[r]) : 0.0f;
                v += __shfl_xor(v, 1);
                v += __shfl_xor(v, 2);
                v += __shfl_xor(v, 4);
                v += __shfl_xor(v, 8);
                if (col == 0)
                    wacc[wave][mt * 16 + hi * 8 + r] += v;
            }
        }
    }
    __syncthreads();
    for (int r = t; r < M_PER_WG; r += 256) {
        float s = 0.0f;
        #pragma unroll
        for (int wv = 0; wv < WAVES; ++wv) s += wacc[wv][r];
        gpart[(size_t)cg * 256 + mb * M_PER_WG + r] = s;
    }
}

// ---------------- kernel 4: deterministic final reduce + margin fixup --------
__global__ void k_loss(const float* __restrict__ gpart,
                       const float* __restrict__ cos_t,
                       float* __restrict__ out) {
    int i = threadIdx.x;                       // one thread per sample
    double s = 0.0;
    for (int g = 0; g < NCG; ++g) s += (double)gpart[(size_t)g * 256 + i];
    float ct = cos_t[i];
    float lt = kScale * (ct - kMargin);
    s += (double)__expf(lt) - (double)__expf(kScale * ct);
    float logp = lt - (float)log(s);
    float li = -fmaxf(logp, -69.07755279f);    // clamp p at 1e-30
    __shared__ float red[256];
    red[i] = li;
    __syncthreads();
    for (int st = 128; st > 0; st >>= 1) {
        if (i < st) red[i] += red[i + st];
        __syncthreads();
    }
    if (i == 0) out[0] = red[0] * (1.0f / 256.0f);
}

// ---------------- host launch ------------------------------------------------
extern "C" void kernel_launch(void* const* d_in, const int* in_sizes, int n_in,
                              void* d_out, int out_size, void* d_ws, size_t ws_size,
                              hipStream_t stream) {
    (void)in_sizes; (void)n_in; (void)out_size; (void)ws_size;
    const float* features = (const float*)d_in[0];
    const int*   targets  = (const int*)d_in[1];
    const float* weight   = (const float*)d_in[2];
    float* out = (float*)d_out;

    char* ws = (char*)d_ws;
    _Float16* Af     = (_Float16*)ws;                       // 256*512*2 = 262144 B
    float*    frnorm = (float*)(ws + 262144);               // 1 KB
    float*    cosT   = (float*)(ws + 263168);               // 1 KB
    float*    gpart  = (float*)(ws + 264192);               // 489*256*4 = 500736 B

    k_norm_feat<<<B_SZ, 256, 0, stream>>>(features, Af, frnorm);
    k_target_cos<<<B_SZ, 256, 0, stream>>>(features, targets, weight, frnorm, cosT);
    k_main<<<2 * NCG, 256, 0, stream>>>(Af, weight, gpart);
    k_loss<<<1, 256, 0, stream>>>(gpart, cosT, out);
}
